// GPTNeoGQASelfAttention_14886356648595
// MI455X (gfx1250) — compile-verified, matrix operand order checked by reading, not run on hardware
//
#include <hip/hip_runtime.h>

// ---- problem constants (from reference) ------------------------------------
#define BATCH   2
#define S_LEN   2048
#define HIDDIM  2048
#define NHEADS  16
#define NKVH    4
#define HDIM    128
#define KVDIM   (NKVH * HDIM)     // 512
#define GQA_G   (NHEADS / NKVH)   // 4
#define TOK     (BATCH * S_LEN)   // 4096

// ---- WMMA types ------------------------------------------------------------
typedef __attribute__((ext_vector_type(16))) __bf16 v16bf;
typedef __attribute__((ext_vector_type(8)))  float  v8f;

union FragU { v16bf v; uint4 q[2]; };

__device__ __forceinline__ unsigned short f2bf(float f) {
  unsigned int u = __float_as_uint(f);
  u += 0x7FFFu + ((u >> 16) & 1u);          // round-to-nearest-even
  return (unsigned short)(u >> 16);
}

__device__ __forceinline__ v8f v8f_zero() {
  v8f z = {0.f, 0.f, 0.f, 0.f, 0.f, 0.f, 0.f, 0.f};
  return z;
}

// Loads one 16-bit WMMA fragment for this lane: elements p[0..7] and p[16..23]
// (A/B 16x32 bf16 layout: K = khalf*8 + 0..7 and 16 + khalf*8 + 0..7).
__device__ __forceinline__ v16bf ld_frag(const unsigned short* p) {
  FragU u;
  u.q[0] = *(const uint4*)(p);
  u.q[1] = *(const uint4*)(p + 16);
  return u.v;
}

__device__ __forceinline__ v8f wmma_bf16(v16bf a, v16bf b, v8f c) {
  // (neg_a, A, neg_b, B, c_mod, C, reuse_a, reuse_b)
  return __builtin_amdgcn_wmma_f32_16x16x32_bf16(false, a, false, b, (short)0, c,
                                                 false, false);
}

// ---- CDNA5 async global->LDS copy (ASYNCcnt tracked, no VGPR data) ---------
// GVS mode: mem addr = SADDR(64) + VADDR(32) ; LDS dst addr from VGPR.
__device__ __forceinline__ void async_ld_b128(unsigned lds_addr, unsigned voff,
                                              const void* sbase) {
  asm volatile("global_load_async_to_lds_b128 %0, %1, %2"
               :
               : "v"(lds_addr), "v"(voff), "s"(sbase)
               : "memory");
}
__device__ __forceinline__ void wait_async_le8() {
  asm volatile("s_wait_asynccnt 0x8" ::: "memory");
}
__device__ __forceinline__ void wait_async_0() {
  asm volatile("s_wait_asynccnt 0x0" ::: "memory");
}
// Low 32 bits of a generic pointer to LDS == wave-relative LDS byte address
// (ISA 10.2).
__device__ __forceinline__ unsigned lds_addr32(const void* p) {
  return (unsigned)(unsigned long long)p;
}

// ---- f32 -> bf16 conversion ------------------------------------------------
__global__ void cvt_f32_bf16(const float* __restrict__ in,
                             unsigned short* __restrict__ out, int n) {
  int i = blockIdx.x * blockDim.x + threadIdx.x;
  if (i < n) out[i] = f2bf(in[i]);
}

// ---- NT GEMM: C[M,N] = A[M,K] * W[N,K]^T  (bf16 in, f32 accum) -------------
// 128x128 block tile, BK=64, double-buffered LDS filled by async-to-LDS DMA.
// 8 waves: (wm in 0..1 rows of 64) x (wn in 0..3 cols of 32); each wave holds
// a 4x2 grid of 16x16 WMMA accumulators and runs 16 WMMAs per K tile.
template <bool OUTF32, bool BIAS>
__global__ __launch_bounds__(256)
void gemm_nt_bf16(const unsigned short* __restrict__ A,
                  const unsigned short* __restrict__ W,
                  void* __restrict__ Cout, const float* __restrict__ bias,
                  int M, int N, int K) {
  constexpr int BK = 64, LD = BK + 8;     // pad: row stride 144 B (16B-aligned)
  __shared__ unsigned short sA[2][128][LD];
  __shared__ unsigned short sB[2][128][LD];

  const int tid  = threadIdx.x;
  const int lane = tid & 31, wave = tid >> 5;
  const int wm = (wave & 1) * 64, wn = (wave >> 1) * 32;
  const int m0 = blockIdx.y * 128, n0 = blockIdx.x * 128;
  const int frow  = lane & 15;            // A/B fragment row (M or N index)
  const int khalf = (lane >> 4) * 8;      // K-half select
  const int crow  = (lane >> 4) * 8;      // C fragment row base
  const int ccol  = lane & 15;            // C fragment col

  v8f acc[4][2];
#pragma unroll
  for (int i = 0; i < 4; ++i)
#pragma unroll
    for (int j = 0; j < 2; ++j) acc[i][j] = v8f_zero();

  // tile fill mapping: 2 threads per row, 32 bf16 (4 x b128) per thread
  const int lr  = tid >> 1;               // 0..127
  const int lcs = (tid & 1) * 32;         // 0 / 32
  const unsigned offA = (unsigned)(((m0 + lr) * K + lcs) * 2);
  const unsigned offW = (unsigned)(((n0 + lr) * K + lcs) * 2);

  auto issue_tile = [&](int bi, int k0) {
    const unsigned la = lds_addr32(&sA[bi][lr][lcs]);
    const unsigned lb = lds_addr32(&sB[bi][lr][lcs]);
    const unsigned ka = offA + (unsigned)(k0 * 2);
    const unsigned kb = offW + (unsigned)(k0 * 2);
#pragma unroll
    for (int i = 0; i < 4; ++i) {
      async_ld_b128(la + i * 16, ka + i * 16, A);
      async_ld_b128(lb + i * 16, kb + i * 16, W);
    }
  };

  const int NK = K / BK;
  issue_tile(0, 0);
  for (int kt = 0; kt < NK; ++kt) {
    const int bi = kt & 1;
    if (kt + 1 < NK) {
      issue_tile(bi ^ 1, (kt + 1) * BK);  // prefetch next tile (stays in flight)
      wait_async_le8();                   // tile kt done (in-order completion)
    } else {
      wait_async_0();
    }
    __syncthreads();

#pragma unroll
    for (int kc = 0; kc < 2; ++kc) {
      v16bf af[4], bfr[2];
#pragma unroll
      for (int i = 0; i < 4; ++i)
        af[i] = ld_frag(&sA[bi][wm + i * 16 + frow][kc * 32 + khalf]);
#pragma unroll
      for (int j = 0; j < 2; ++j)
        bfr[j] = ld_frag(&sB[bi][wn + j * 16 + frow][kc * 32 + khalf]);
#pragma unroll
      for (int i = 0; i < 4; ++i)
#pragma unroll
        for (int j = 0; j < 2; ++j)
          acc[i][j] = wmma_bf16(af[i], bfr[j], acc[i][j]);
    }
    __syncthreads();   // all waves done with buffer bi before it is refilled
  }

#pragma unroll
  for (int i = 0; i < 4; ++i)
#pragma unroll
    for (int j = 0; j < 2; ++j)
#pragma unroll
      for (int g = 0; g < 8; ++g) {
        const int mm = m0 + wm + i * 16 + crow + g;
        const int nn = n0 + wn + j * 16 + ccol;
        float v = acc[i][j][g];
        if (BIAS) v += bias[nn];
        if (OUTF32) ((float*)Cout)[(size_t)mm * N + nn] = v;
        else ((unsigned short*)Cout)[(size_t)mm * N + nn] = f2bf(v);
      }
}

// ---- Flash attention (causal, no 1/sqrt(d) scale, GQA) ---------------------
// Grid (S/128, NH, B). 128 q rows per block; wave w owns rows [w*16, w*16+16)
// with a full 16x128 f32 accumulator. Streams 32 kv rows per iteration.
// K tile is filled with async-to-LDS DMA; V is transposed through registers.
__global__ __launch_bounds__(256)
void flash_attn_bf16(const unsigned short* __restrict__ Q,
                     const unsigned short* __restrict__ Kb,
                     const unsigned short* __restrict__ Vb,
                     unsigned short* __restrict__ O) {
  constexpr int AQ = 128, AKV = 32;
  constexpr int LDK = HDIM + 8, LDV = AKV + 8, LDP = AKV + 8;
  __shared__ unsigned short sK[AKV][LDK];     // [kv][d]
  __shared__ unsigned short sVt[HDIM][LDV];   // [d][kv]  (transposed V)
  __shared__ unsigned short sP[8][16][LDP];   // per-wave P relayout buffer

  const int q0  = blockIdx.x * AQ;
  const int h   = blockIdx.y;
  const int b   = blockIdx.z;
  const int kvh = h / GQA_G;
  const int tid = threadIdx.x;
  const int lane = tid & 31, wave = tid >> 5;
  const int frow = lane & 15, khalf = (lane >> 4) * 8;
  const int crow = (lane >> 4) * 8, ccol = lane & 15;
  const int qw = q0 + wave * 16;              // this wave's first q row

  // Q fragments: 16 rows x 128 d = 4 A-frags (K chunks of 32)
  const unsigned short* qptr =
      Q + (size_t)(b * S_LEN + qw + frow) * (NHEADS * HDIM) + h * HDIM;
  v16bf qf[4];
#pragma unroll
  for (int d = 0; d < 4; ++d) qf[d] = ld_frag(qptr + d * 32 + khalf);

  v8f acc[8];
#pragma unroll
  for (int t = 0; t < 8; ++t) acc[t] = v8f_zero();
  float m_i[8], l_i[8];
#pragma unroll
  for (int g = 0; g < 8; ++g) { m_i[g] = -1e30f; l_i[g] = 0.f; }

  const int lr = tid >> 3;                    // 0..31 kv row loaded
  const int lcseg = (tid & 7) * 16;           // 16 d-elements per thread
  const unsigned lkaddr = lds_addr32(&sK[lr][lcseg]);
  const unsigned kvbase =
      (unsigned)(((b * S_LEN + lr) * KVDIM + kvh * HDIM + lcseg) * 2);

  for (int kv0 = 0; kv0 < q0 + AQ; kv0 += AKV) {
    // K tile: async DMA straight into LDS (row-major)
    const unsigned koff = kvbase + (unsigned)(kv0 * KVDIM * 2);
    async_ld_b128(lkaddr, koff, Kb);
    async_ld_b128(lkaddr + 16, koff + 16, Kb);

    // V tile: load 16 elems, transpose through registers into sVt
    const unsigned short* gv =
        Vb + (size_t)(b * S_LEN + kv0 + lr) * KVDIM + kvh * HDIM + lcseg;
    union { uint4 q; unsigned short s[8]; } v0, v1;
    v0.q = *(const uint4*)(gv);
    v1.q = *(const uint4*)(gv + 8);
#pragma unroll
    for (int e = 0; e < 8; ++e) {
      sVt[lcseg + e][lr]     = v0.s[e];
      sVt[lcseg + 8 + e][lr] = v1.s[e];
    }
    wait_async_0();
    __syncthreads();

    // S = Q * K^T : 16 x 32 scores = two 16x16 C frags, 4 chained WMMAs each
    v8f sc[2];
    sc[0] = v8f_zero(); sc[1] = v8f_zero();
#pragma unroll
    for (int n = 0; n < 2; ++n)
#pragma unroll
      for (int d = 0; d < 4; ++d) {
        v16bf kf = ld_frag(&sK[n * 16 + frow][d * 32 + khalf]);
        sc[n] = wmma_bf16(qf[d], kf, sc[n]);
      }

    // online softmax (rows split across 16-lane halves -> xor<=8 reductions)
    float corr[8];
#pragma unroll
    for (int g = 0; g < 8; ++g) {
      const int qq = qw + crow + g;
      float s0 = sc[0][g];
      float s1 = sc[1][g];
      if (kv0 + ccol      > qq) s0 = -1e30f;   // causal mask
      if (kv0 + 16 + ccol > qq) s1 = -1e30f;
      float t = fmaxf(s0, s1);
      t = fmaxf(t, __shfl_xor(t, 1, 32));
      t = fmaxf(t, __shfl_xor(t, 2, 32));
      t = fmaxf(t, __shfl_xor(t, 4, 32));
      t = fmaxf(t, __shfl_xor(t, 8, 32));
      const float mn = fmaxf(m_i[g], t);
      const float p0 = __expf(s0 - mn);
      const float p1 = __expf(s1 - mn);
      float rs = p0 + p1;
      rs += __shfl_xor(rs, 1, 32);
      rs += __shfl_xor(rs, 2, 32);
      rs += __shfl_xor(rs, 4, 32);
      rs += __shfl_xor(rs, 8, 32);
      const float c = __expf(m_i[g] - mn);
      l_i[g] = l_i[g] * c + rs;
      m_i[g] = mn;
      corr[g] = c;
      // C-frag -> A-frag relayout through wave-private LDS slice
      sP[wave][crow + g][ccol]      = f2bf(p0);
      sP[wave][crow + g][16 + ccol] = f2bf(p1);
    }
#pragma unroll
    for (int t = 0; t < 8; ++t)
#pragma unroll
      for (int g = 0; g < 8; ++g) acc[t][g] *= corr[g];

    // O += P * V  (8 d-tiles of 16)
    v16bf pf = ld_frag(&sP[wave][frow][khalf]);
#pragma unroll
    for (int t = 0; t < 8; ++t) {
      v16bf vf = ld_frag(&sVt[t * 16 + frow][khalf]);
      acc[t] = wmma_bf16(pf, vf, acc[t]);
    }
    __syncthreads();
  }

#pragma unroll
  for (int g = 0; g < 8; ++g) l_i[g] = 1.0f / l_i[g];  // diag guarantees l>0
#pragma unroll
  for (int t = 0; t < 8; ++t)
#pragma unroll
    for (int g = 0; g < 8; ++g) {
      const int qq = qw + crow + g;
      const int d  = t * 16 + ccol;
      O[(size_t)(b * S_LEN + qq) * (NHEADS * HDIM) + h * HDIM + d] =
          f2bf(acc[t][g] * l_i[g]);
    }
}

// ---- host launch -----------------------------------------------------------
extern "C" void kernel_launch(void* const* d_in, const int* in_sizes, int n_in,
                              void* d_out, int out_size, void* d_ws,
                              size_t ws_size, hipStream_t stream) {
  (void)in_sizes; (void)n_in; (void)out_size; (void)ws_size;
  const float* hs = (const float*)d_in[0];
  const float* Wq = (const float*)d_in[1];
  const float* Wk = (const float*)d_in[2];
  const float* Wv = (const float*)d_in[3];
  const float* Wo = (const float*)d_in[4];
  const float* bo = (const float*)d_in[5];

  unsigned short* ws = (unsigned short*)d_ws;
  size_t off = 0;
  unsigned short* hsb = ws + off; off += (size_t)TOK * HIDDIM;
  unsigned short* wqb = ws + off; off += (size_t)HIDDIM * HIDDIM;
  unsigned short* wkb = ws + off; off += (size_t)KVDIM * HIDDIM;
  unsigned short* wvb = ws + off; off += (size_t)KVDIM * HIDDIM;
  unsigned short* wob = ws + off; off += (size_t)HIDDIM * HIDDIM;
  unsigned short* qb  = ws + off; off += (size_t)TOK * HIDDIM;
  unsigned short* kb  = ws + off; off += (size_t)TOK * KVDIM;
  unsigned short* vb  = ws + off; off += (size_t)TOK * KVDIM;
  unsigned short* ab  = ws + off; off += (size_t)TOK * HIDDIM;

  auto cvt = [&](const float* src, unsigned short* dst, int n) {
    cvt_f32_bf16<<<dim3((n + 255) / 256), dim3(256), 0, stream>>>(src, dst, n);
  };
  cvt(hs, hsb, TOK * HIDDIM);
  cvt(Wq, wqb, HIDDIM * HIDDIM);
  cvt(Wk, wkb, KVDIM * HIDDIM);
  cvt(Wv, wvb, KVDIM * HIDDIM);
  cvt(Wo, wob, HIDDIM * HIDDIM);

  dim3 blk(256);
  gemm_nt_bf16<false, false><<<dim3(HIDDIM / 128, TOK / 128), blk, 0, stream>>>(
      hsb, wqb, qb, nullptr, TOK, HIDDIM, HIDDIM);
  gemm_nt_bf16<false, false><<<dim3(KVDIM / 128, TOK / 128), blk, 0, stream>>>(
      hsb, wkb, kb, nullptr, TOK, KVDIM, HIDDIM);
  gemm_nt_bf16<false, false><<<dim3(KVDIM / 128, TOK / 128), blk, 0, stream>>>(
      hsb, wvb, vb, nullptr, TOK, KVDIM, HIDDIM);

  flash_attn_bf16<<<dim3(S_LEN / 128, NHEADS, BATCH), blk, 0, stream>>>(
      qb, kb, vb, ab);

  gemm_nt_bf16<true, true><<<dim3(HIDDIM / 128, TOK / 128), blk, 0, stream>>>(
      ab, wob, d_out, bo, TOK, HIDDIM, HIDDIM);
}
